// Attention_43233140802225
// MI455X (gfx1250) — compile-verified, matrix operand order checked by reading, not run on hardware
//
#include <hip/hip_runtime.h>

// ---------------------------------------------------------------------------
// CDNA5 (gfx1250) attention: bf16 WMMA (f32 accum) everywhere.
//   Q = x Wq^T, K = x Wk^T, V = x Wv^T      (GEMM, fp32 in -> bf16 out)
//   flash attention per (b, h)              (QK^T, online softmax, PV)
//   out = attn Wo^T                         (GEMM, bf16 in -> fp32 out)
// GEMMs are double-buffered in LDS; bf16->bf16 staging uses the gfx1250
// async global->LDS path (ASYNCcnt) when the builtin is available.
// ---------------------------------------------------------------------------

typedef __attribute__((ext_vector_type(8)))  float   v8f;
typedef __attribute__((ext_vector_type(16))) __bf16  v16bf;
typedef __attribute__((ext_vector_type(8)))  __bf16  v8bf;
typedef __attribute__((ext_vector_type(4)))  int     v4i;

#define DIM   2048
#define SEQ   2048
#define BATCH 2
#define HEADS 16
#define HDIM  128
#define MROWS (BATCH * SEQ)   // 4096

#if defined(__has_builtin)
#if __has_builtin(__builtin_amdgcn_global_load_async_to_lds_b128)
#define ASYNC_CP 1
#endif
#endif

#ifdef ASYNC_CP
typedef __attribute__((address_space(1))) v4i* g4p;  // global int4*
typedef __attribute__((address_space(3))) v4i* l4p;  // LDS int4*
// Per-lane 16B async copy global -> LDS (tracked by ASYNCcnt, no VGPR data).
__device__ __forceinline__ void async_cp16(const void* g, void* l) {
  __builtin_amdgcn_global_load_async_to_lds_b128((g4p)(g), (l4p)(l), 0, 0);
}
__device__ __forceinline__ void wait_async0() {
#if __has_builtin(__builtin_amdgcn_s_wait_asynccnt)
  __builtin_amdgcn_s_wait_asynccnt(0);
#else
  asm volatile("s_wait_asynccnt 0x0" ::: "memory");
#endif
}
#endif

__device__ __forceinline__ v8f wmma_bf16(v16bf a, v16bf b, v8f c) {
  // (neg_a, A, neg_b, B, c_mod, C, reuse_a, reuse_b)
  return __builtin_amdgcn_wmma_f32_16x16x32_bf16(false, a, false, b, (short)0, c,
                                                 false, false);
}

// Load one 16-bf16 A/B fragment from a row-major bf16 row.
// Hardware layout (16-bit, 16x32): lane L -> row L%16; K elems at
// half*8..half*8+7 and 16+half*8..16+half*8+7 (half = L/16), each 16B chunk.
__device__ __forceinline__ v16bf frag_ld(const __bf16* rowbase, int half) {
  const v8bf lo = *(const v8bf*)(rowbase + half * 8);
  const v8bf hi = *(const v8bf*)(rowbase + 16 + half * 8);
  return __builtin_shufflevector(lo, hi, 0, 1, 2, 3, 4, 5, 6, 7,
                                 8, 9, 10, 11, 12, 13, 14, 15);
}

__device__ __forceinline__ v8bf cvt8(const float4 a, const float4 b) {
  v8bf r;
  r[0] = (__bf16)a.x; r[1] = (__bf16)a.y; r[2] = (__bf16)a.z; r[3] = (__bf16)a.w;
  r[4] = (__bf16)b.x; r[5] = (__bf16)b.y; r[6] = (__bf16)b.z; r[7] = (__bf16)b.w;
  return r;
}

// ---------------------------------------------------------------------------
// GEMM: C[M,N] = A[M,K] * W[N,K]^T.  A fp32 or bf16 (row-major, K contig),
// W fp32 row-major [N][K], C bf16 or fp32 row-major.
// 256 threads = 8 waves; 128x128 tile; waves 4(M) x 2(N); wave = 32x64.
// Double-buffered LDS: stage step+1 while WMMAs consume step.
// ---------------------------------------------------------------------------
template <bool A_BF16, bool OUT_F32>
__global__ __launch_bounds__(256) void gemm_xWT(const void* __restrict__ Aptr,
                                                const float* __restrict__ W,
                                                void* __restrict__ Out,
                                                int M, int N, int K) {
  __shared__ __align__(16) __bf16 As[2][128][40];  // +8 pad: conflict-free frags
  __shared__ __align__(16) __bf16 Bs[2][128][40];

  const int t    = threadIdx.x;
  const int lane = t & 31;
  const int wave = t >> 5;
  const int half = lane >> 4;
  const int r    = lane & 15;
  const int mw   = wave >> 1;  // 0..3
  const int nw   = wave & 1;   // 0..1
  const int tileM = blockIdx.y * 128;
  const int tileN = blockIdx.x * 128;
  const int KSTEPS = K / 32;

  // Stage one 128x32 K-slab of A and W (as bf16) into LDS buffer `buf`.
  auto stage = [&](int k0, int buf) {
#pragma unroll
    for (int it = 0; it < 2; ++it) {
      const int c   = it * 256 + t;   // 0..511
      const int row = c >> 2;         // 0..127
      const int col = (c & 3) * 8;    // 0,8,16,24
      if constexpr (A_BF16) {
        const __bf16* gp =
            (const __bf16*)Aptr + (size_t)(tileM + row) * K + k0 + col;
#ifdef ASYNC_CP
        async_cp16(gp, &As[buf][row][col]);
#else
        *(v8bf*)&As[buf][row][col] = *(const v8bf*)gp;
#endif
      } else {
        const float* Af = (const float*)Aptr;
        const float4 f0 = *(const float4*)(Af + (size_t)(tileM + row) * K + k0 + col);
        const float4 f1 = *(const float4*)(Af + (size_t)(tileM + row) * K + k0 + col + 4);
        *(v8bf*)&As[buf][row][col] = cvt8(f0, f1);
      }
      const float4 g0 = *(const float4*)(W + (size_t)(tileN + row) * K + k0 + col);
      const float4 g1 = *(const float4*)(W + (size_t)(tileN + row) * K + k0 + col + 4);
      *(v8bf*)&Bs[buf][row][col] = cvt8(g0, g1);
    }
  };

  v8f acc[2][4];
#pragma unroll
  for (int i = 0; i < 2; ++i)
#pragma unroll
    for (int j = 0; j < 4; ++j) acc[i][j] = (v8f){0, 0, 0, 0, 0, 0, 0, 0};

  stage(0, 0);
#ifdef ASYNC_CP
  if constexpr (A_BF16) wait_async0();
#endif
  __syncthreads();

  for (int step = 0; step < KSTEPS; ++step) {
    const int cur = step & 1;
    if (step + 1 < KSTEPS) stage((step + 1) * 32, cur ^ 1);

    v16bf af[2], bf[4];
#pragma unroll
    for (int mt = 0; mt < 2; ++mt)
      af[mt] = frag_ld(&As[cur][mw * 32 + mt * 16 + r][0], half);
#pragma unroll
    for (int nt = 0; nt < 4; ++nt)
      bf[nt] = frag_ld(&Bs[cur][nw * 64 + nt * 16 + r][0], half);
#pragma unroll
    for (int mt = 0; mt < 2; ++mt)
#pragma unroll
      for (int nt = 0; nt < 4; ++nt)
        acc[mt][nt] = wmma_bf16(af[mt], bf[nt], acc[mt][nt]);

#ifdef ASYNC_CP
    if constexpr (A_BF16) {
      if (step + 1 < KSTEPS) wait_async0();
    }
#endif
    __syncthreads();
  }

  // C/D layout: lane holds n = r, VGPR v holds m = v + 8*half.
#pragma unroll
  for (int mt = 0; mt < 2; ++mt)
#pragma unroll
    for (int nt = 0; nt < 4; ++nt)
#pragma unroll
      for (int v = 0; v < 8; ++v) {
        const int m = tileM + mw * 32 + mt * 16 + v + 8 * half;
        const int n = tileN + nw * 64 + nt * 16 + r;
        const float val = acc[mt][nt][v];
        if constexpr (OUT_F32)
          ((float*)Out)[(size_t)m * N + n] = val;
        else
          ((__bf16*)Out)[(size_t)m * N + n] = (__bf16)val;
      }
}

// ---------------------------------------------------------------------------
// Flash attention: grid (S/64, HEADS, BATCH), 128 threads = 4 waves.
// Wave w owns query rows q0 + w*16 .. +15. Key blocks of 32, causal-bounded.
// K tile staged via async global->LDS; V staged transposed through VGPRs.
// ---------------------------------------------------------------------------
__global__ __launch_bounds__(128) void flash_attn(const __bf16* __restrict__ Q,
                                                  const __bf16* __restrict__ Kg,
                                                  const __bf16* __restrict__ Vg,
                                                  __bf16* __restrict__ Og) {
  __shared__ __align__(16) __bf16 Ks[32][136];   // key-major, d contig (+8 pad)
  __shared__ __align__(16) __bf16 Vt[128][40];   // d-major, key contig (+8 pad)
  __shared__ __align__(16) __bf16 Ps[4][16][40]; // per-wave P tile 16q x 32k

  const int t    = threadIdx.x;
  const int lane = t & 31;
  const int w    = t >> 5;
  const int half = lane >> 4;
  const int r    = lane & 15;
  const int b    = blockIdx.z;
  const int h    = blockIdx.y;
  const int q0   = blockIdx.x * 64;
  const int qb   = q0 + w * 16;

  const size_t headoff = (size_t)b * SEQ * DIM + (size_t)h * HDIM;
  const __bf16* Qh = Q + headoff;
  const __bf16* Kh = Kg + headoff;
  const __bf16* Vh = Vg + headoff;

  // Resident Q fragments: 4 K-steps of 32 over head_dim=128.
  v16bf qf[4];
#pragma unroll
  for (int ks = 0; ks < 4; ++ks) {
    const __bf16* rowp = Qh + (size_t)(qb + r) * DIM + ks * 32;
    qf[ks] = frag_ld(rowp, half);
  }

  float m_i[8], l_i[8];
  v8f o[8];
#pragma unroll
  for (int v = 0; v < 8; ++v) { m_i[v] = -1e30f; l_i[v] = 0.f; }
#pragma unroll
  for (int dt = 0; dt < 8; ++dt) o[dt] = (v8f){0, 0, 0, 0, 0, 0, 0, 0};

  const float sc  = 0.08838834764831845f;  // 1/sqrt(128)
  const float l2e = 1.44269504088896f;
  const int nkb = q0 / 32 + 2;  // cover keys 0 .. q0+63 (causal)

  for (int kb = 0; kb < nkb; ++kb) {
    const int kstart = kb * 32;
    // Stage K (row-major, async copy) and V (transposed) tiles: 32 x 128.
#pragma unroll
    for (int it = 0; it < 4; ++it) {
      const int idx = it * 128 + t;   // 0..511 chunks of 8 bf16
      const int row = idx >> 4;       // 0..31 (key)
      const int col = (idx & 15) * 8; // 0..120 (d)
      const __bf16* kp = Kh + (size_t)(kstart + row) * DIM + col;
#ifdef ASYNC_CP
      async_cp16(kp, &Ks[row][col]);
#else
      *(v8bf*)&Ks[row][col] = *(const v8bf*)kp;
#endif
      const v8bf vv = *(const v8bf*)(Vh + (size_t)(kstart + row) * DIM + col);
#pragma unroll
      for (int j = 0; j < 8; ++j) Vt[col + j][row] = vv[j];
    }
#ifdef ASYNC_CP
    wait_async0();
#endif
    __syncthreads();

    // scores = Q K^T for two 16-key subtiles.
    v8f sacc[2];
    sacc[0] = (v8f){0, 0, 0, 0, 0, 0, 0, 0};
    sacc[1] = (v8f){0, 0, 0, 0, 0, 0, 0, 0};
#pragma unroll
    for (int ks = 0; ks < 4; ++ks) {
#pragma unroll
      for (int kt = 0; kt < 2; ++kt) {
        const v16bf kf = frag_ld(&Ks[kt * 16 + r][ks * 32], half);
        sacc[kt] = wmma_bf16(qf[ks], kf, sacc[kt]);
      }
    }

    // Online softmax; accumulator row q = v + 8*half lives across the 16
    // lanes of one half -> butterfly over xor masks 1,2,4,8.
    float osc[8];
#pragma unroll
    for (int v = 0; v < 8; ++v) {
      const int qi = qb + v + 8 * half;
      float s0 = sacc[0][v] * sc;
      float s1 = sacc[1][v] * sc;
      if (kstart + r > qi)      s0 = -1e30f;   // causal mask
      if (kstart + 16 + r > qi) s1 = -1e30f;
      float bm = fmaxf(s0, s1);
#pragma unroll
      for (int msk = 1; msk <= 8; msk <<= 1) bm = fmaxf(bm, __shfl_xor(bm, msk, 32));
      const float mnew  = fmaxf(m_i[v], bm);
      const float scale = exp2f((m_i[v] - mnew) * l2e);
      const float p0 = exp2f((s0 - mnew) * l2e);
      const float p1 = exp2f((s1 - mnew) * l2e);
      float rs = p0 + p1;
#pragma unroll
      for (int msk = 1; msk <= 8; msk <<= 1) rs += __shfl_xor(rs, msk, 32);
      l_i[v] = l_i[v] * scale + rs;
      m_i[v] = mnew;
      osc[v] = scale;
      Ps[w][v + 8 * half][r]      = (__bf16)p0;
      Ps[w][v + 8 * half][16 + r] = (__bf16)p1;
    }
#pragma unroll
    for (int dt = 0; dt < 8; ++dt)
#pragma unroll
      for (int v = 0; v < 8; ++v) o[dt][v] *= osc[v];
    __syncthreads();

    // O += P V  (K-dim = 32 keys; 8 d-subtiles of 16).
    const v16bf pf = frag_ld(&Ps[w][r][0], half);
#pragma unroll
    for (int dt = 0; dt < 8; ++dt) {
      const v16bf vf = frag_ld(&Vt[dt * 16 + r][0], half);
      o[dt] = wmma_bf16(pf, vf, o[dt]);
    }
    __syncthreads();
  }

  // Normalize and write bf16 attention output at [b][s][h*128 + d].
#pragma unroll
  for (int dt = 0; dt < 8; ++dt)
#pragma unroll
    for (int v = 0; v < 8; ++v) {
      const int s = qb + v + 8 * half;
      const int d = dt * 16 + r;
      const float val = o[dt][v] / l_i[v];
      Og[((size_t)b * SEQ + s) * DIM + (size_t)h * HDIM + d] = (__bf16)val;
    }
}

// ---------------------------------------------------------------------------
extern "C" void kernel_launch(void* const* d_in, const int* in_sizes, int n_in,
                              void* d_out, int out_size, void* d_ws, size_t ws_size,
                              hipStream_t stream) {
  const float* x  = (const float*)d_in[0];
  // d_in[1] = mask (causal mask reproduced analytically in-kernel)
  const float* wq = (const float*)d_in[2];
  const float* wk = (const float*)d_in[3];
  const float* wv = (const float*)d_in[4];
  const float* wo = (const float*)d_in[5];

  const size_t tensor_elems = (size_t)MROWS * DIM;  // 4096 * 2048
  __bf16* Qb = (__bf16*)d_ws;
  __bf16* Kb = Qb + tensor_elems;
  __bf16* Vb = Kb + tensor_elems;
  __bf16* Ab = Vb + tensor_elems;

  dim3 gg(DIM / 128, MROWS / 128);  // (16, 32)
  gemm_xWT<false, false><<<gg, 256, 0, stream>>>(x, wq, Qb, MROWS, DIM, DIM);
  gemm_xWT<false, false><<<gg, 256, 0, stream>>>(x, wk, Kb, MROWS, DIM, DIM);
  gemm_xWT<false, false><<<gg, 256, 0, stream>>>(x, wv, Vb, MROWS, DIM, DIM);

  dim3 ga(SEQ / 64, HEADS, BATCH);  // (32, 16, 2)
  flash_attn<<<ga, 128, 0, stream>>>(Qb, Kb, Vb, Ab);

  gemm_xWT<true, true><<<gg, 256, 0, stream>>>(Ab, wo, (float*)d_out, MROWS, DIM, DIM);
}
